// PVConv_22514218566078
// MI455X (gfx1250) — matrix operand order repaired and checked by reading, not compiled, not run on hardware
//
#include <hip/hip_runtime.h>
#include <hip/hip_bf16.h>
#include <math.h>

typedef __attribute__((ext_vector_type(16))) __bf16 v16bf;
typedef __attribute__((ext_vector_type(8)))  float  v8f;

#define RVOX 32
#define R3   32768
#define NPTS 65536
#define CIN1 64
#define COUT 128
#define BATCH 4

// ---------------- workspace layout (bytes) ----------------
#define OFF_GRID  ((size_t)0)                       // 4*64*32768 f32  = 33,554,432
#define OFF_H1    (OFF_GRID + 33554432u)            // 4*128*32768 f32 = 67,108,864
#define OFF_H2    (OFF_H1   + 67108864u)            // 67,108,864
#define OFF_NC    (OFF_H2   + 67108864u)            // 4*3*65536 f32   =  3,145,728
#define OFF_CNT   (OFF_NC   + 3145728u)             // 4*32768 f32     =    524,288
#define OFF_W1    (OFF_CNT  + 524288u)              // 1728*128 bf16   =    442,368
#define OFF_W2    (OFF_W1   + 442368u)              // 3456*128 bf16   =    884,736
#define OFF_WP    (OFF_W2   + 884736u)              // 64*128 bf16     =     16,384
#define OFF_STATS (OFF_WP   + 16384u)               // small f32 scratch

// ---------------- utility: zero f32 range ----------------
__global__ void zero_f32_k(float* __restrict__ p, int n) {
    for (int i = blockIdx.x * blockDim.x + threadIdx.x; i < n;
         i += gridDim.x * blockDim.x)
        p[i] = 0.f;
}

// ---------------- weight pack: [O][K] f32 -> WMMA B-fragment-major bf16 -------
// Packed layout: Bp[ ((k/32)*128 + o)*32 + (k%32) ]
// => a lane's 16 B-fragment elements (K = k0 + khalf + 0..15, fixed column o)
//    are 32 contiguous bytes -> two global_load_b128 per chunk.
__global__ void wpack_k(const float* __restrict__ w, __bf16* __restrict__ o,
                        int KT, int total) {
    for (int t = blockIdx.x * blockDim.x + threadIdx.x; t < total;
         t += gridDim.x * blockDim.x) {
        int oc = t / KT;
        int k  = t - oc * KT;
        o[(((k >> 5) * COUT + oc) << 5) + (k & 31)] = (__bf16)w[t];
    }
}

// ---------------- per-batch coordinate mean ----------------
__global__ void coord_mean_k(const float* __restrict__ coords,
                             float* __restrict__ stats) {
    int b = blockIdx.x;
    int tid = threadIdx.x;
    float s0 = 0.f, s1 = 0.f, s2 = 0.f;
    for (int n = tid; n < NPTS; n += 256) {
        s0 += coords[((b * 3 + 0) << 16) + n];
        s1 += coords[((b * 3 + 1) << 16) + n];
        s2 += coords[((b * 3 + 2) << 16) + n];
    }
    __shared__ float r0[256], r1[256], r2[256];
    r0[tid] = s0; r1[tid] = s1; r2[tid] = s2;
    __syncthreads();
    for (int off = 128; off > 0; off >>= 1) {
        if (tid < off) { r0[tid] += r0[tid+off]; r1[tid] += r1[tid+off]; r2[tid] += r2[tid+off]; }
        __syncthreads();
    }
    if (tid == 0) {
        const float inv = 1.f / (float)NPTS;
        stats[b * 3 + 0] = r0[0] * inv;
        stats[b * 3 + 1] = r1[0] * inv;
        stats[b * 3 + 2] = r2[0] * inv;
    }
}

// ---------------- per-batch max ||centered||, denom = 2*max ----------------
__global__ void coord_maxnorm_k(const float* __restrict__ coords,
                                float* __restrict__ stats) {
    int b = blockIdx.x;
    int tid = threadIdx.x;
    float m0 = stats[b * 3 + 0], m1 = stats[b * 3 + 1], m2 = stats[b * 3 + 2];
    float mx = 0.f;
    for (int n = tid; n < NPTS; n += 256) {
        float dx = coords[((b * 3 + 0) << 16) + n] - m0;
        float dy = coords[((b * 3 + 1) << 16) + n] - m1;
        float dz = coords[((b * 3 + 2) << 16) + n] - m2;
        mx = fmaxf(mx, sqrtf(dx * dx + dy * dy + dz * dz));
    }
    __shared__ float r[256];
    r[tid] = mx;
    __syncthreads();
    for (int off = 128; off > 0; off >>= 1) {
        if (tid < off) r[tid] = fmaxf(r[tid], r[tid + off]);
        __syncthreads();
    }
    if (tid == 0) stats[12 + b] = r[0] * 2.f;
}

// ---------------- voxelization scatter (sum + counts) ----------------
__global__ void voxel_scatter_k(const float* __restrict__ coords,
                                const float* __restrict__ feats,
                                float* __restrict__ nc,
                                float* __restrict__ grid,
                                float* __restrict__ cnt,
                                const float* __restrict__ stats) {
    int t = blockIdx.x * blockDim.x + threadIdx.x;   // 0 .. 262143
    if (t >= BATCH * NPTS) return;
    int b = t >> 16, n = t & (NPTS - 1);
    float inv = 1.f / stats[12 + b];
    int vi[3];
#pragma unroll
    for (int d = 0; d < 3; ++d) {
        float c = coords[((b * 3 + d) << 16) + n] - stats[b * 3 + d];
        float v = c * inv + 0.5f;
        v = fminf(fmaxf(v * (float)RVOX, 0.f), (float)(RVOX - 1));
        nc[((b * 3 + d) << 16) + n] = v;
        vi[d] = (int)rintf(v);            // round-half-to-even, matches jnp.round
    }
    int flat = (vi[0] * RVOX + vi[1]) * RVOX + vi[2];
    atomicAdd(&cnt[(b << 15) + flat], 1.f);
#pragma unroll 4
    for (int c = 0; c < CIN1; ++c)
        atomicAdd(&grid[((b * CIN1 + c) << 15) + flat],
                  feats[((b * CIN1 + c) << 16) + n]);
}

// ---------------- sum -> mean ----------------
__global__ void voxel_finalize_k(float* __restrict__ grid,
                                 const float* __restrict__ cnt) {
    const int total = BATCH * CIN1 * R3;             // 8,388,608
    for (int i = blockIdx.x * blockDim.x + threadIdx.x; i < total;
         i += gridDim.x * blockDim.x) {
        int v = i & (R3 - 1);
        int b = i >> 21;                             // 64*32768 = 2^21
        grid[i] /= fmaxf(cnt[(b << 15) + v], 1.f);
    }
}

// ---------------- implicit-GEMM 3x3x3 conv via bf16 WMMA ----------------
// Workgroup = 8 waves; tile = 64 positions (M) x 128 out channels (N).
// Each wave: 4 accumulators (4 M-subtiles x its 16-channel slab); B fragment
// loaded once per K-chunk (32 contiguous bytes, fragment-major packed panel)
// and reused across the 4 WMMAs.
template <int CIN>
__global__ void conv3d_wmma_k(const float* __restrict__ grid,
                              const __bf16* __restrict__ w,
                              const float* __restrict__ bias,
                              float* __restrict__ out) {
    const int KTOT = CIN * 27;
    __shared__ __bf16 Alds[64 * 32];
    int tid  = threadIdx.x;
    int wave = tid >> 5;
    int lane = tid & 31;
    int mt = blockIdx.x;                 // B * 512 M-tiles
    int b  = mt >> 9;
    int m0 = (mt & 511) << 6;            // base flattened voxel position (64 wide)

    // per-thread staging coordinates: fixed kk column, rows m_s + 8*j
    int kk_s = tid & 31;
    int m_s  = tid >> 5;

    v8f acc0 = {}, acc1 = {}, acc2 = {}, acc3 = {};
    for (int k0 = 0; k0 < KTOT; k0 += 32) {
        // hoisted im2col decomposition (fixed per chunk per thread)
        int k  = k0 + kk_s;
        int ci = k / 27, kidx = k - ci * 27;
        int dz = kidx % 3 - 1;
        int dy = (kidx / 3) % 3 - 1;
        int dx = kidx / 9 - 1;
        const float* gp = grid + ((b * CIN + ci) << 15);

        __syncthreads();                 // protect LDS from previous iter readers
#pragma unroll
        for (int j = 0; j < 8; ++j) {
            int m   = m_s + (j << 3);
            int pos = m0 + m;
            int z = pos & 31, y = (pos >> 5) & 31, x = pos >> 10;
            int xx = x + dx, yy = y + dy, zz = z + dz;
            float v = 0.f;
            if ((unsigned)xx < 32u && (unsigned)yy < 32u && (unsigned)zz < 32u)
                v = gp[(xx << 10) + (yy << 5) + zz];
            Alds[(m << 5) + kk_s] = (__bf16)v;
        }
        __syncthreads();

        // B fragment: 32 contiguous bytes from the packed panel
        int o = (wave << 4) + (lane & 15);
        v16bf bfrag = *(const v16bf*)(
            w + ((((k0 >> 5) * COUT + o) << 5) + ((lane < 16) ? 0 : 16)));

        // A fragments in contiguous k order (K = khalf + 0..15), matching B
        int mrow  = lane & 15;
        int khalf = (lane < 16) ? 0 : 16;
        v16bf a0, a1, a2, a3;
#pragma unroll
        for (int i = 0; i < 16; ++i) {
            int kk = i + khalf;
            a0[i] = Alds[((mrow)      << 5) + kk];
            a1[i] = Alds[((16 + mrow) << 5) + kk];
            a2[i] = Alds[((32 + mrow) << 5) + kk];
            a3[i] = Alds[((48 + mrow) << 5) + kk];
        }
        acc0 = __builtin_amdgcn_wmma_f32_16x16x32_bf16(
            false, a0, false, bfrag, (short)0, acc0, false, false);
        acc1 = __builtin_amdgcn_wmma_f32_16x16x32_bf16(
            false, a1, false, bfrag, (short)0, acc1, false, false);
        acc2 = __builtin_amdgcn_wmma_f32_16x16x32_bf16(
            false, a2, false, bfrag, (short)0, acc2, false, false);
        acc3 = __builtin_amdgcn_wmma_f32_16x16x32_bf16(
            false, a3, false, bfrag, (short)0, acc3, false, false);
    }

    // D layout: lane -> channel, VGPR r -> position (contiguous f32 stores)
    int o = (wave << 4) + (lane & 15);
    float bv = bias[o];
    float* op = out + ((b * COUT + o) << 15) + m0 + ((lane < 16) ? 0 : 8);
#pragma unroll
    for (int r = 0; r < 8; ++r) {
        op[r]      = acc0[r] + bv;
        op[16 + r] = acc1[r] + bv;
        op[32 + r] = acc2[r] + bv;
        op[48 + r] = acc3[r] + bv;
    }
}

// ---------------- point branch 1x1 GEMM via bf16 WMMA (M=64 tiles) ----------
__global__ void point_wmma_k(const float* __restrict__ feats,
                             const __bf16* __restrict__ w,
                             const float* __restrict__ bias,
                             float* __restrict__ out) {
    __shared__ __bf16 Alds[64 * 32];
    int tid  = threadIdx.x;
    int wave = tid >> 5;
    int lane = tid & 31;
    int mt = blockIdx.x;                 // B * 1024 M-tiles
    int b  = mt >> 10;
    int m0 = (mt & 1023) << 6;

    int kk_s = tid & 31;
    int m_s  = tid >> 5;

    v8f acc0 = {}, acc1 = {}, acc2 = {}, acc3 = {};
    for (int k0 = 0; k0 < CIN1; k0 += 32) {
        const float* fp = feats + ((b * CIN1 + k0 + kk_s) << 16) + m0;
        __syncthreads();
#pragma unroll
        for (int j = 0; j < 8; ++j) {
            int m = m_s + (j << 3);
            Alds[(m << 5) + kk_s] = (__bf16)fp[m];
        }
        __syncthreads();

        int o = (wave << 4) + (lane & 15);
        v16bf bfrag = *(const v16bf*)(
            w + ((((k0 >> 5) * COUT + o) << 5) + ((lane < 16) ? 0 : 16)));

        int mrow  = lane & 15;
        int khalf = (lane < 16) ? 0 : 16;
        v16bf a0, a1, a2, a3;
#pragma unroll
        for (int i = 0; i < 16; ++i) {
            int kk = i + khalf;
            a0[i] = Alds[((mrow)      << 5) + kk];
            a1[i] = Alds[((16 + mrow) << 5) + kk];
            a2[i] = Alds[((32 + mrow) << 5) + kk];
            a3[i] = Alds[((48 + mrow) << 5) + kk];
        }
        acc0 = __builtin_amdgcn_wmma_f32_16x16x32_bf16(
            false, a0, false, bfrag, (short)0, acc0, false, false);
        acc1 = __builtin_amdgcn_wmma_f32_16x16x32_bf16(
            false, a1, false, bfrag, (short)0, acc1, false, false);
        acc2 = __builtin_amdgcn_wmma_f32_16x16x32_bf16(
            false, a2, false, bfrag, (short)0, acc2, false, false);
        acc3 = __builtin_amdgcn_wmma_f32_16x16x32_bf16(
            false, a3, false, bfrag, (short)0, acc3, false, false);
    }
    int o = (wave << 4) + (lane & 15);
    float bv = bias[o];
    float* op = out + ((b * COUT + o) << 16) + m0 + ((lane < 16) ? 0 : 8);
#pragma unroll
    for (int r = 0; r < 8; ++r) {
        op[r]      = acc0[r] + bv;
        op[16 + r] = acc1[r] + bv;
        op[32 + r] = acc2[r] + bv;
        op[48 + r] = acc3[r] + bv;
    }
}

// ---------------- GroupNorm stats (one block per (b, group)) ----------------
__global__ void gn_stats_k(const float* __restrict__ x, float* __restrict__ stats,
                           int S, int cpg) {
    int bg = blockIdx.x;                 // b * 8 + g
    int tid = threadIdx.x;
    size_t base = (size_t)bg * (size_t)cpg * (size_t)S;
    size_t n = (size_t)cpg * (size_t)S;
    float s = 0.f, q = 0.f;
    for (size_t i = tid; i < n; i += 256) {
        float v = x[base + i];
        s += v; q += v * v;
    }
    __shared__ float rs[256], rq[256];
    rs[tid] = s; rq[tid] = q;
    __syncthreads();
    for (int off = 128; off > 0; off >>= 1) {
        if (tid < off) { rs[tid] += rs[tid + off]; rq[tid] += rq[tid + off]; }
        __syncthreads();
    }
    if (tid == 0) {
        float inv = 1.f / (float)n;
        float mu  = rs[0] * inv;
        float var = rq[0] * inv - mu * mu;
        stats[bg * 2 + 0] = mu;
        stats[bg * 2 + 1] = rsqrtf(var + 1e-5f);
    }
}

// ---------------- fused GN affine + SiLU (in place) ----------------
__global__ void gn_silu_k(float* __restrict__ x, const float* __restrict__ stats,
                          const float* __restrict__ scale,
                          const float* __restrict__ bias,
                          int logS, int total) {
    for (int i = blockIdx.x * blockDim.x + threadIdx.x; i < total;
         i += gridDim.x * blockDim.x) {
        int c = (i >> logS) & (COUT - 1);
        int b = i >> (logS + 7);
        int bg = b * 8 + (c >> 4);       // 16 channels per group
        float v = (x[i] - stats[bg * 2]) * stats[bg * 2 + 1] * scale[c] + bias[c];
        x[i] = v / (1.f + expf(-v));
    }
}

// ---------------- trilinear devoxelize, accumulate into out ----------------
__global__ void devox_add_k(const float* __restrict__ h,
                            const float* __restrict__ nc,
                            float* __restrict__ out) {
    const int total = BATCH * COUT * NPTS;           // 33,554,432
    for (int i = blockIdx.x * blockDim.x + threadIdx.x; i < total;
         i += gridDim.x * blockDim.x) {
        int n = i & (NPTS - 1);
        int c = (i >> 16) & (COUT - 1);
        int b = i >> 23;
        float fx = nc[((b * 3 + 0) << 16) + n];
        float fy = nc[((b * 3 + 1) << 16) + n];
        float fz = nc[((b * 3 + 2) << 16) + n];
        int lx = (int)floorf(fx), ly = (int)floorf(fy), lz = (int)floorf(fz);
        float tx = fx - (float)lx, ty = fy - (float)ly, tz = fz - (float)lz;
        int hx = min(lx + 1, RVOX - 1), hy = min(ly + 1, RVOX - 1), hz = min(lz + 1, RVOX - 1);
        const float* g = h + ((b * COUT + c) << 15);
        float a =
            g[(lx * 32 + ly) * 32 + lz] * (1.f - tx) * (1.f - ty) * (1.f - tz) +
            g[(lx * 32 + ly) * 32 + hz] * (1.f - tx) * (1.f - ty) * tz +
            g[(lx * 32 + hy) * 32 + lz] * (1.f - tx) * ty * (1.f - tz) +
            g[(lx * 32 + hy) * 32 + hz] * (1.f - tx) * ty * tz +
            g[(hx * 32 + ly) * 32 + lz] * tx * (1.f - ty) * (1.f - tz) +
            g[(hx * 32 + ly) * 32 + hz] * tx * (1.f - ty) * tz +
            g[(hx * 32 + hy) * 32 + lz] * tx * ty * (1.f - tz) +
            g[(hx * 32 + hy) * 32 + hz] * tx * ty * tz;
        out[i] += a;
    }
}

extern "C" void kernel_launch(void* const* d_in, const int* in_sizes, int n_in,
                              void* d_out, int out_size, void* d_ws, size_t ws_size,
                              hipStream_t stream) {
    const float* coords = (const float*)d_in[0];
    const float* feats  = (const float*)d_in[1];
    const float* c1w = (const float*)d_in[2];
    const float* c1b = (const float*)d_in[3];
    const float* g1s = (const float*)d_in[4];
    const float* g1b = (const float*)d_in[5];
    const float* c2w = (const float*)d_in[6];
    const float* c2b = (const float*)d_in[7];
    const float* g2s = (const float*)d_in[8];
    const float* g2b = (const float*)d_in[9];
    const float* pw  = (const float*)d_in[10];
    const float* pb  = (const float*)d_in[11];
    const float* pgs = (const float*)d_in[12];
    const float* pgb = (const float*)d_in[13];
    float* out = (float*)d_out;

    char* ws = (char*)d_ws;
    float*  grid  = (float*)(ws + OFF_GRID);
    float*  h1    = (float*)(ws + OFF_H1);
    float*  h2    = (float*)(ws + OFF_H2);
    float*  nc    = (float*)(ws + OFF_NC);
    float*  cnt   = (float*)(ws + OFF_CNT);
    __bf16* w1b   = (__bf16*)(ws + OFF_W1);
    __bf16* w2b   = (__bf16*)(ws + OFF_W2);
    __bf16* wpb   = (__bf16*)(ws + OFF_WP);
    float*  stats = (float*)(ws + OFF_STATS); // [0..11] mean, [12..15] denom
    float*  gnst  = stats + 16;               // 32 (mu, rstd) pairs, reused

    // zero scatter targets + stats
    zero_f32_k<<<4096, 256, 0, stream>>>(grid, BATCH * CIN1 * R3);
    zero_f32_k<<<512,  256, 0, stream>>>(cnt, BATCH * R3);
    zero_f32_k<<<1,    256, 0, stream>>>(stats, 128);

    // weight panels: [O][K] f32 -> fragment-major bf16
    wpack_k<<<864,  256, 0, stream>>>(c1w, w1b, CIN1 * 27, COUT * CIN1 * 27);
    wpack_k<<<1728, 256, 0, stream>>>(c2w, w2b, COUT * 27, COUT * COUT * 27);
    wpack_k<<<32,   256, 0, stream>>>(pw,  wpb, CIN1,      COUT * CIN1);

    // voxelization
    coord_mean_k<<<BATCH, 256, 0, stream>>>(coords, stats);
    coord_maxnorm_k<<<BATCH, 256, 0, stream>>>(coords, stats);
    voxel_scatter_k<<<(BATCH * NPTS) / 256, 256, 0, stream>>>(coords, feats, nc,
                                                              grid, cnt, stats);
    voxel_finalize_k<<<4096, 256, 0, stream>>>(grid, cnt);

    // voxel branch: conv1 -> GN -> SiLU -> conv2 -> GN -> SiLU
    conv3d_wmma_k<CIN1><<<BATCH * 512, 256, 0, stream>>>(grid, w1b, c1b, h1);
    gn_stats_k<<<32, 256, 0, stream>>>(h1, gnst, R3, 16);
    gn_silu_k<<<8192, 256, 0, stream>>>(h1, gnst, g1s, g1b, 15, BATCH * COUT * R3);
    conv3d_wmma_k<COUT><<<BATCH * 512, 256, 0, stream>>>(h1, w2b, c2b, h2);
    gn_stats_k<<<32, 256, 0, stream>>>(h2, gnst, R3, 16);
    gn_silu_k<<<8192, 256, 0, stream>>>(h2, gnst, g2s, g2b, 15, BATCH * COUT * R3);

    // point branch directly into d_out
    point_wmma_k<<<BATCH * 1024, 256, 0, stream>>>(feats, wpb, pb, out);
    gn_stats_k<<<32, 256, 0, stream>>>(out, gnst, NPTS, 16);
    gn_silu_k<<<16384, 256, 0, stream>>>(out, gnst, pgs, pgb, 16,
                                         BATCH * COUT * NPTS);

    // devoxelize voxel branch and accumulate
    devox_add_k<<<16384, 256, 0, stream>>>(h2, nc, out);
}
